// GeneNetwork_84456236908940
// MI455X (gfx1250) — matrix operand order repaired and checked by reading, not compiled, not run on hardware
//
#include <hip/hip_runtime.h>
#include <cstdint>

typedef __attribute__((ext_vector_type(16))) _Float16 v16h;
typedef __attribute__((ext_vector_type(8)))  _Float16 v8h;
typedef __attribute__((ext_vector_type(8)))  float    v8f;

#define DIMC 64
#define EPSV 1e-5f
#define RED_BLOCKS 1024
#define GEMM_BLOCKS 256

// ---------------- utility kernels ----------------

__global__ void k_zero_f32(float* __restrict__ p, long long n) {
  long long i = (long long)blockIdx.x * blockDim.x + threadIdx.x;
  if (i < n) p[i] = 0.f;
}

__global__ void k_edge_deg(const long long* __restrict__ dst, float* __restrict__ deg, int E) {
  int e = blockIdx.x * blockDim.x + threadIdx.x;
  if (e < E) atomicAdd(&deg[dst[e]], 1.0f);
}

__global__ void k_node_dinv(const float* __restrict__ deg, float* __restrict__ dinv, int N) {
  int i = blockIdx.x * blockDim.x + threadIdx.x;
  if (i < N) {
    float d = deg[i];
    dinv[i] = (d > 0.f) ? rsqrtf(fmaxf(d, 1.f)) : 0.f;
  }
}

__global__ void k_edge_norm(const long long* __restrict__ src, const long long* __restrict__ dst,
                            const float* __restrict__ dinv, float* __restrict__ norm, int E) {
  int e = blockIdx.x * blockDim.x + threadIdx.x;
  if (e < E) norm[e] = dinv[src[e]] * dinv[dst[e]];
}

// one hop: out[dst] += h[src] * norm  (edge-parallel, float4 per thread)
__global__ void k_prop_scatter(const float* __restrict__ h, const long long* __restrict__ src,
                               const long long* __restrict__ dst, const float* __restrict__ norm,
                               float* __restrict__ out, long long total) {
  long long t = (long long)blockIdx.x * blockDim.x + threadIdx.x;
  if (t >= total) return;
  long long e = t >> 4;
  int q = (int)(t & 15) * 4;
  long long s = src[e], d = dst[e];
  float nm = norm[e];
  float4 v = *(const float4*)(h + s * DIMC + q);
  float* o = out + d * DIMC + q;
  atomicAdd(o + 0, v.x * nm);
  atomicAdd(o + 1, v.y * nm);
  atomicAdd(o + 2, v.z * nm);
  atomicAdd(o + 3, v.w * nm);
}

// ---------------- WMMA GEMM ----------------
// C[M, NT*16] = (beta ? C : 0) + A[M, 0:64 of lda] @ W[64, NT*16]  (+bias, optional ELU)
// K is fixed at 64 (two passes handle K=128). A row stride = lda.
// block = 128 threads (4 waves); each wave owns one 16-row tile per grid-stride step.
// All B fragments live in registers; inner loop is pure VMEM + WMMA.
template <int NT>
__global__ __attribute__((amdgpu_waves_per_eu(1)))
void __launch_bounds__(128) k_gemm_wmma(const float* __restrict__ A, int lda,
                                        const float* __restrict__ W,
                                        const float* __restrict__ bias, float* __restrict__ C,
                                        int mTiles, int beta, int act) {
  constexpr int K    = 64;
  constexpr int NOUT = NT * 16;
  constexpr int KP   = K + 8;              // padded column stride (halves) -> bank-conflict-free
  constexpr int NCK  = K / 32;
  __shared__ _Float16 sWt[NOUT * KP];      // W transposed: [n][k], k-contiguous

  const int tid  = threadIdx.x;
  const int lane = tid & 31;
  const int wv   = tid >> 5;
  const int m16  = lane & 15;
  const int hi   = lane >> 4;

  // stage W transposed into LDS (coalesced global reads); W rows are NOUT apart
  for (int i = tid; i < K * NOUT; i += blockDim.x) {
    int k = i / NOUT, n = i % NOUT;
    sWt[n * KP + k] = (_Float16)W[i];
  }
  __syncthreads();

  // preload ALL B fragments into registers (kept live across the tile loop)
  v16h bfrag[NT][NCK];
  #pragma unroll
  for (int nt = 0; nt < NT; ++nt) {
    #pragma unroll
    for (int ck = 0; ck < NCK; ++ck) {
      const _Float16* bp = &sWt[(nt * 16 + m16) * KP + ck * 32 + 8 * hi];
      v8h b0 = *(const v8h*)(bp);
      v8h b1 = *(const v8h*)(bp + 16);
      bfrag[nt][ck] = __builtin_shufflevector(b0, b1, 0, 1, 2, 3, 4, 5, 6, 7,
                                                      8, 9, 10, 11, 12, 13, 14, 15);
    }
  }

  for (int tile = blockIdx.x * 4 + wv; tile < mTiles; tile += gridDim.x * 4) {
    const int rowBase = tile * 16;

    v8f c[NT];
    if (beta) {
      #pragma unroll
      for (int nt = 0; nt < NT; ++nt)
        #pragma unroll
        for (int v = 0; v < 8; ++v)
          c[nt][v] = C[(long long)(rowBase + v + 8 * hi) * NOUT + nt * 16 + m16];
    } else {
      #pragma unroll
      for (int nt = 0; nt < NT; ++nt)
        #pragma unroll
        for (int v = 0; v < 8; ++v) c[nt][v] = 0.f;
    }

    const long long rowOff = (long long)(rowBase + m16) * lda;

    #pragma unroll
    for (int ck = 0; ck < NCK; ++ck) {
      // A fragment: lane (m16, hi) needs k = [8hi, 8hi+8) and [16+8hi, 16+8hi+8)
      const float* ap = A + rowOff + ck * 32 + 8 * hi;
      float4 p0 = *(const float4*)(ap);
      float4 p1 = *(const float4*)(ap + 4);
      float4 p2 = *(const float4*)(ap + 16);
      float4 p3 = *(const float4*)(ap + 20);
      v16h a;
      a[0]  = (_Float16)p0.x; a[1]  = (_Float16)p0.y; a[2]  = (_Float16)p0.z; a[3]  = (_Float16)p0.w;
      a[4]  = (_Float16)p1.x; a[5]  = (_Float16)p1.y; a[6]  = (_Float16)p1.z; a[7]  = (_Float16)p1.w;
      a[8]  = (_Float16)p2.x; a[9]  = (_Float16)p2.y; a[10] = (_Float16)p2.z; a[11] = (_Float16)p2.w;
      a[12] = (_Float16)p3.x; a[13] = (_Float16)p3.y; a[14] = (_Float16)p3.z; a[15] = (_Float16)p3.w;

      #pragma unroll
      for (int nt = 0; nt < NT; ++nt)
        c[nt] = __builtin_amdgcn_wmma_f32_16x16x32_f16(false, a, false, bfrag[nt][ck],
                                                       (short)0, c[nt], false, false);
    }

    #pragma unroll
    for (int nt = 0; nt < NT; ++nt) {
      const int col = nt * 16 + m16;
      #pragma unroll
      for (int v = 0; v < 8; ++v) {
        int row = rowBase + v + 8 * hi;
        float x = c[nt][v];
        if (bias) x += bias[col];
        if (act)  x = (x > 0.f) ? x : (expf(x) - 1.f);
        C[(long long)row * NOUT + col] = x;
      }
    }
  }
}

// ---------------- LayerNorm (graph mode) ----------------

__global__ void k_reduce_partial(const float* __restrict__ x, long long n, float* __restrict__ part) {
  __shared__ float s1[256], s2[256];
  float a = 0.f, b = 0.f;
  for (long long i = (long long)blockIdx.x * blockDim.x + threadIdx.x; i < n;
       i += (long long)gridDim.x * blockDim.x) {
    float v = x[i];
    a += v; b += v * v;
  }
  s1[threadIdx.x] = a; s2[threadIdx.x] = b;
  __syncthreads();
  for (int off = 128; off > 0; off >>= 1) {
    if ((int)threadIdx.x < off) {
      s1[threadIdx.x] += s1[threadIdx.x + off];
      s2[threadIdx.x] += s2[threadIdx.x + off];
    }
    __syncthreads();
  }
  if (threadIdx.x == 0) {
    part[2 * blockIdx.x]     = s1[0];
    part[2 * blockIdx.x + 1] = s2[0];
  }
}

__global__ void k_reduce_final(const float* __restrict__ part, int nparts, long long n,
                               float* __restrict__ stats) {
  __shared__ float s1[256], s2[256];
  float a = 0.f, b = 0.f;
  for (int i = threadIdx.x; i < nparts; i += 256) {
    a += part[2 * i];
    b += part[2 * i + 1];
  }
  s1[threadIdx.x] = a; s2[threadIdx.x] = b;
  __syncthreads();
  for (int off = 128; off > 0; off >>= 1) {
    if ((int)threadIdx.x < off) {
      s1[threadIdx.x] += s1[threadIdx.x + off];
      s2[threadIdx.x] += s2[threadIdx.x + off];
    }
    __syncthreads();
  }
  if (threadIdx.x == 0) {
    float mean = s1[0] / (float)n;
    float var  = fmaxf(s2[0] / (float)n - mean * mean, 0.f);
    stats[0] = mean;
    stats[1] = 1.f / (sqrtf(var) + EPSV);
  }
}

__global__ void k_apply_ln(const float* __restrict__ in, const float* __restrict__ stats,
                           const float* __restrict__ w, const float* __restrict__ b,
                           float* __restrict__ out, long long total) {
  long long t = (long long)blockIdx.x * blockDim.x + threadIdx.x;
  if (t >= total) return;
  int col = (int)(t & 63);
  out[t] = (in[t] - stats[0]) * stats[1] * w[col] + b[col];
}

// ---------------- final 32 -> 1 matvec ----------------

__global__ void k_matvec_out(const float* __restrict__ h, const float* __restrict__ w,
                             const float* __restrict__ b, float* __restrict__ out, int N) {
  int i = blockIdx.x * blockDim.x + threadIdx.x;
  if (i >= N) return;
  float s = b[0];
  #pragma unroll
  for (int j = 0; j < 32; ++j) s += h[(long long)i * 32 + j] * w[j];
  out[i] = s;
}

// ---------------- host orchestration ----------------

static inline int cdiv(long long a, int b) { return (int)((a + b - 1) / b); }

extern "C" void kernel_launch(void* const* d_in, const int* in_sizes, int n_in,
                              void* d_out, int out_size, void* d_ws, size_t ws_size,
                              hipStream_t stream) {
  const int IN_DIM = 128, D = 64, KHOPS = 3, NL = 2;
  const int N = in_sizes[0] / IN_DIM;   // 100000 (multiple of 16)
  const int E = in_sizes[1] / 2;

  const float*     x       = (const float*)d_in[0];
  const long long* ei      = (const long long*)d_in[1];
  const long long* src     = ei;
  const long long* dst     = ei + E;
  const float*     pre_w0  = (const float*)d_in[2];
  const float*     pre_b0  = (const float*)d_in[3];
  const float*     pre_w1  = (const float*)d_in[4];
  const float*     pre_b1  = (const float*)d_in[5];
  const float*     mp_lins = (const float*)d_in[6];   // [NL, K+1, D, D]
  const float*     mp_bias = (const float*)d_in[7];   // [NL, D]
  const float*     ln_w    = (const float*)d_in[8];
  const float*     ln_b    = (const float*)d_in[9];
  const float*     post_w0 = (const float*)d_in[10];
  const float*     post_b0 = (const float*)d_in[11];
  const float*     post_w1 = (const float*)d_in[12];
  const float*     post_b1 = (const float*)d_in[13];
  const float*     post_w2 = (const float*)d_in[14];
  const float*     post_b2 = (const float*)d_in[15];
  float*           out     = (float*)d_out;

  // workspace layout (floats), 256-float aligned
  float* ws = (float*)d_ws;
  long long off = 0;
  auto alloc = [&](long long nflt) {
    float* p = ws + off;
    off += (nflt + 255) & ~255LL;
    return p;
  };
  float* deg   = alloc(N);
  float* dinv  = alloc(N);
  float* norm  = alloc(E);
  float* h     = alloc((long long)N * D);
  float* acc   = alloc((long long)N * D);
  float* hkA   = alloc((long long)N * D);
  float* hkB   = alloc((long long)N * D);
  float* part  = alloc(2 * RED_BLOCKS);
  float* stats = alloc(2);
  (void)ws_size; (void)n_in; (void)out_size;

  const long long ND = (long long)N * D;
  const long long EF = (long long)E * 16;  // float4-granular scatter threads
  const int mT = N / 16;

  // --- gcn_norm ---
  k_zero_f32<<<cdiv(N, 256), 256, 0, stream>>>(deg, N);
  k_edge_deg<<<cdiv(E, 256), 256, 0, stream>>>(dst, deg, E);
  k_node_dinv<<<cdiv(N, 256), 256, 0, stream>>>(deg, dinv, N);
  k_edge_norm<<<cdiv(E, 256), 256, 0, stream>>>(src, dst, dinv, norm, E);

  // --- pre_mp: ELU(x@W0+b0) as two K=64 passes, then ELU(.@W1+b1) ---
  k_gemm_wmma<4><<<GEMM_BLOCKS, 128, 0, stream>>>(x,      IN_DIM, pre_w0,             nullptr, acc, mT, 0, 0);
  k_gemm_wmma<4><<<GEMM_BLOCKS, 128, 0, stream>>>(x + 64, IN_DIM, pre_w0 + 64 * D,    pre_b0,  acc, mT, 1, 1);
  k_gemm_wmma<4><<<GEMM_BLOCKS, 128, 0, stream>>>(acc,    D,      pre_w1,             pre_b1,  h,   mT, 0, 1);

  // --- mp layers: TAGConv -> ELU -> graph LayerNorm ---
  for (int l = 0; l < NL; ++l) {
    const float* W = mp_lins + (long long)l * (KHOPS + 1) * D * D;

    // hop 0
    k_gemm_wmma<4><<<GEMM_BLOCKS, 128, 0, stream>>>(h, D, W, nullptr, acc, mT, 0, 0);
    // hop 1
    k_zero_f32<<<cdiv(ND, 256), 256, 0, stream>>>(hkA, ND);
    k_prop_scatter<<<cdiv(EF, 256), 256, 0, stream>>>(h, src, dst, norm, hkA, EF);
    k_gemm_wmma<4><<<GEMM_BLOCKS, 128, 0, stream>>>(hkA, D, W + 1 * D * D, nullptr, acc, mT, 1, 0);
    // hop 2
    k_zero_f32<<<cdiv(ND, 256), 256, 0, stream>>>(hkB, ND);
    k_prop_scatter<<<cdiv(EF, 256), 256, 0, stream>>>(hkA, src, dst, norm, hkB, EF);
    k_gemm_wmma<4><<<GEMM_BLOCKS, 128, 0, stream>>>(hkB, D, W + 2 * D * D, nullptr, acc, mT, 1, 0);
    // hop 3 (+bias, +ELU fused)
    k_zero_f32<<<cdiv(ND, 256), 256, 0, stream>>>(hkA, ND);
    k_prop_scatter<<<cdiv(EF, 256), 256, 0, stream>>>(hkB, src, dst, norm, hkA, EF);
    k_gemm_wmma<4><<<GEMM_BLOCKS, 128, 0, stream>>>(hkA, D, W + 3 * D * D,
                                                    mp_bias + (long long)l * D, acc, mT, 1, 1);

    // graph-mode LayerNorm over all N*D elements
    k_reduce_partial<<<RED_BLOCKS, 256, 0, stream>>>(acc, ND, part);
    k_reduce_final<<<1, 256, 0, stream>>>(part, RED_BLOCKS, ND, stats);
    k_apply_ln<<<cdiv(ND, 256), 256, 0, stream>>>(acc, stats, ln_w + (long long)l * D,
                                                  ln_b + (long long)l * D, h, ND);
  }

  // --- post_mp ---
  k_gemm_wmma<4><<<GEMM_BLOCKS, 128, 0, stream>>>(h,   D, post_w0, post_b0, acc, mT, 0, 1);
  k_gemm_wmma<2><<<GEMM_BLOCKS, 128, 0, stream>>>(acc, D, post_w1, post_b1, hkA, mT, 0, 1);
  k_matvec_out<<<cdiv(N, 256), 256, 0, stream>>>(hkA, post_w2, post_b2, out, N);
}
